// DRAW_17497696763951
// MI455X (gfx1250) — matrix operand order repaired and checked by reading, not compiled, not run on hardware
//
#include <hip/hip_runtime.h>
#include <cstdint>
#include <cstddef>

// Problem sizes (match reference)
#define B_   4096
#define X_   1024
#define H_   256
#define Z_   10
#define T_   10
#define G_   1024      // 4*H
#define KENC 2560      // X + X + H + H   ([x | x_hat | h_dec | h_enc])
#define KDEC 288       // H + 32 (z padded from 10 to 32)

typedef __bf16 bf16_t;
typedef __attribute__((ext_vector_type(8)))  __bf16 v8bf;
typedef __attribute__((ext_vector_type(16))) __bf16 v16bf;
typedef __attribute__((ext_vector_type(8)))  float  v8f;

__device__ __forceinline__ float sigmoidf_(float v) { return 1.f / (1.f + __expf(-v)); }

// ---------------------------------------------------------------------------
// CDNA5 async global->LDS copy (ASYNCcnt-tracked), 16B per lane.
// lds = wave-relative LDS byte address (low 32 bits of flat-cast shared ptr)
// ---------------------------------------------------------------------------
__device__ __forceinline__ void async_copy_b128(uint32_t lds, const void* gptr) {
    asm volatile("global_load_async_to_lds_b128 %0, %1, off"
                 :: "v"(lds), "v"((uint64_t)(uintptr_t)gptr)
                 : "memory");
}
__device__ __forceinline__ void wait_async_le5() {
    asm volatile("s_wait_asynccnt 0x5" ::: "memory");
}
__device__ __forceinline__ void wait_async_le0() {
    asm volatile("s_wait_asynccnt 0x0" ::: "memory");
}
__device__ __forceinline__ void sched_fence() {
#if __has_builtin(__builtin_amdgcn_sched_barrier)
    __builtin_amdgcn_sched_barrier(0);
#endif
}

// ---------------------------------------------------------------------------
// bf16 WMMA GEMM (compile-time K):
//   C[M,N](f32) = A[M,K](bf16) * W[N,K](bf16)^T  (+C if ACC) + bias
// Block tile 256x64, K-step 32, 256 threads (8 waves).
// Each wave owns a 32x64 strip: 2 M sub-tiles x 4 N tiles -> 8 WMMA / k-step.
// Double-buffered LDS; tiles DMA'd via global_load_async_to_lds_b128 (5/tile);
// explicit prologue / steady-state / epilogue software pipeline.
// ---------------------------------------------------------------------------
#define TM 256
#define TN 64
#define TK 32
#define LDSP 40   // padded LDS row stride (bf16): conflict-free b128 accesses

template<int K, bool ACC>
__global__ __launch_bounds__(256, 2)
void k_gemm_bf16(const bf16_t* __restrict__ A, int lda,
                 const bf16_t* __restrict__ W, int ldw,
                 float* __restrict__ C, int ldc,
                 const float* __restrict__ bias)
{
    constexpr int NITER = K / TK;

    __shared__ __align__(16) bf16_t As[2][TM][LDSP];
    __shared__ __align__(16) bf16_t Bs[2][TN][LDSP];

    const int tid  = threadIdx.x;
    const int wv   = tid >> 5;       // wave 0..7 -> 32-row M strip
    const int lane = tid & 31;
    const int hp   = lane >> 4;      // lane half
    const int l16  = lane & 15;
    const int m0 = blockIdx.y * TM;
    const int n0 = blockIdx.x * TN;

    v8f acc[2][4];
#pragma unroll
    for (int s = 0; s < 2; ++s)
#pragma unroll
        for (int i = 0; i < 4; ++i)
#pragma unroll
            for (int e = 0; e < 8; ++e) acc[s][i][e] = 0.f;

    // cooperative tile DMA: A: 256 rows x 32 cols, 1 thread/row (4x16B each)
    // B: 64 rows x 32 cols, 4 threads/row (1x16B each)
    const int brow = tid >> 2;
    const int bcol = (tid & 3) * 8;

    const bf16_t* Abase = A + (size_t)(m0 + tid) * lda;
    const bf16_t* Wbase = W + (size_t)(n0 + brow) * ldw + bcol;

    uint32_t ldsA[2], ldsB[2];
    ldsA[0] = (uint32_t)(uintptr_t)&As[0][tid][0];
    ldsA[1] = (uint32_t)(uintptr_t)&As[1][tid][0];
    ldsB[0] = (uint32_t)(uintptr_t)&Bs[0][brow][bcol];
    ldsB[1] = (uint32_t)(uintptr_t)&Bs[1][brow][bcol];

    auto prefetch = [&](int tile, int buf) {
        const int k0 = tile * TK;
        async_copy_b128(ldsA[buf],      Abase + k0);
        async_copy_b128(ldsA[buf] + 16, Abase + k0 + 8);
        async_copy_b128(ldsA[buf] + 32, Abase + k0 + 16);
        async_copy_b128(ldsA[buf] + 48, Abase + k0 + 24);
        async_copy_b128(ldsB[buf],      Wbase + k0);
    };

    auto compute = [&](int buf) {
        // A fragments: row = wv*32 + s*16 + l16; K chunks at hp*8 / 16+hp*8
        v16bf afr[2];
#pragma unroll
        for (int s = 0; s < 2; ++s) {
            const int am = wv * 32 + s * 16 + l16;
            v8bf a0 = *(const v8bf*)&As[buf][am][hp * 8];
            v8bf a1 = *(const v8bf*)&As[buf][am][16 + hp * 8];
            afr[s] = __builtin_shufflevector(a0, a1,
                0,1,2,3,4,5,6,7,8,9,10,11,12,13,14,15);
        }
        // B fragments: col = nt*16 + l16; K chunks at hp*8 / 16+hp*8
        v16bf bfr[4];
#pragma unroll
        for (int nt = 0; nt < 4; ++nt) {
            const int bn = nt * 16 + l16;
            v8bf b0 = *(const v8bf*)&Bs[buf][bn][hp * 8];
            v8bf b1 = *(const v8bf*)&Bs[buf][bn][16 + hp * 8];
            bfr[nt] = __builtin_shufflevector(b0, b1,
                0,1,2,3,4,5,6,7,8,9,10,11,12,13,14,15);
        }
        sched_fence();   // keep all ds_loads ahead of the WMMA burst
#pragma unroll
        for (int s = 0; s < 2; ++s)
#pragma unroll
            for (int nt = 0; nt < 4; ++nt)
                acc[s][nt] = __builtin_amdgcn_wmma_f32_16x16x32_bf16(
                    false, afr[s], false, bfr[nt], (short)0, acc[s][nt],
                    false, false);
    };

    // ---- software pipeline ----
    prefetch(0, 0);
    if constexpr (NITER > 1) prefetch(1, 1);

#pragma unroll 2
    for (int it = 0; it < NITER - 2; ++it) {
        wait_async_le5();        // in-order: oldest 5 (this buffer) have landed
        __syncthreads();
        compute(it & 1);
        __syncthreads();         // everyone done reading before overwrite
        prefetch(it + 2, it & 1);
    }
    if constexpr (NITER >= 2) {  // tile NITER-2, buf (NITER-2)&1 == NITER&1
        wait_async_le5();
        __syncthreads();
        compute(NITER & 1);
    }
    wait_async_le0();            // tile NITER-1
    __syncthreads();
    compute((NITER - 1) & 1);

    // C layout: VGPR v -> row hp*8+v, col l16
#pragma unroll
    for (int s = 0; s < 2; ++s)
#pragma unroll
        for (int nt = 0; nt < 4; ++nt) {
            const int col = n0 + nt * 16 + l16;
            const float bv = bias[col];
#pragma unroll
            for (int v = 0; v < 8; ++v) {
                const int row = m0 + wv * 32 + s * 16 + hp * 8 + v;
                float* p = C + (size_t)row * ldc + col;
                float val = acc[s][nt][v] + bv;
                if constexpr (ACC) val += *p;
                *p = val;
            }
        }
}

// ---------------------------------------------------------------------------
// Weight / state preparation kernels
// ---------------------------------------------------------------------------
__global__ void k_zero_f32(float* p, int n) {
    int i = blockIdx.x * blockDim.x + threadIdx.x;
    if (i < n) p[i] = 0.f;
}
__global__ void k_zero_bf16(bf16_t* p, int n) {
    int i = blockIdx.x * blockDim.x + threadIdx.x;
    if (i < n) p[i] = (bf16_t)0.f;
}
__global__ void k_f32_to_bf16(const float* __restrict__ in, bf16_t* __restrict__ out, int n) {
    int i = blockIdx.x * blockDim.x + threadIdx.x;
    if (i < n) out[i] = (bf16_t)in[i];
}
__global__ void k_add_f32(const float* a, const float* b, float* out, int n) {
    int i = blockIdx.x * blockDim.x + threadIdx.x;
    if (i < n) out[i] = a[i] + b[i];
}
// Wenc[n, 0:2304] = W_ih_enc[n,:], Wenc[n, 2304:2560] = W_hh_enc[n,:]
__global__ void k_build_wenc(const float* __restrict__ Wih, const float* __restrict__ Whh,
                             bf16_t* __restrict__ out) {
    int idx = blockIdx.x * blockDim.x + threadIdx.x;
    if (idx >= G_ * KENC) return;
    int n = idx / KENC, k = idx % KENC;
    float v = (k < 2304) ? Wih[n * 2304 + k] : Whh[n * H_ + (k - 2304)];
    out[idx] = (bf16_t)v;
}
// Wdec[n, 0:256] = W_hh_dec, Wdec[n, 256:266] = W_ih_dec, rest zero
__global__ void k_build_wdec(const float* __restrict__ Whh, const float* __restrict__ Wih,
                             bf16_t* __restrict__ out) {
    int idx = blockIdx.x * blockDim.x + threadIdx.x;
    if (idx >= G_ * KDEC) return;
    int n = idx / KDEC, k = idx % KDEC;
    float v = (k < H_) ? Whh[n * H_ + k] : ((k < H_ + Z_) ? Wih[n * Z_ + (k - H_)] : 0.f);
    out[idx] = (bf16_t)v;
}
// att init: cols 0:1024 = bf16(x) (constant over steps), cols 1024:2560 zeroed
__global__ void k_init_att(const float* __restrict__ x, bf16_t* __restrict__ att) {
    int idx = blockIdx.x * blockDim.x + threadIdx.x;
    if (idx >= B_ * KENC) return;
    int m = idx / KENC, j = idx % KENC;
    att[idx] = (j < X_) ? (bf16_t)x[m * X_ + j] : (bf16_t)0.f;
}

// ---------------------------------------------------------------------------
// Per-step pointwise kernels
// ---------------------------------------------------------------------------
// att[:, 1024:2048] = bf16(x - sigmoid(canvas))
__global__ void k_att_xhat(const float* __restrict__ x, const float* __restrict__ canvas,
                           bf16_t* __restrict__ att) {
    int idx = blockIdx.x * blockDim.x + threadIdx.x;
    if (idx >= B_ * X_) return;
    int m = idx / X_, j = idx % X_;
    float v = x[idx] - sigmoidf_(canvas[idx]);
    att[(size_t)m * KENC + X_ + j] = (bf16_t)v;
}
// encoder LSTM cell: consumes gates (B x 4H, [i f g o]); updates c_enc;
// writes h_enc (f32 for var head) and att[:, 2304:2560] (bf16 for next GEMM)
__global__ void k_enc_cell(const float* __restrict__ gates, float* __restrict__ c_enc,
                           float* __restrict__ h_enc, bf16_t* __restrict__ att) {
    int idx = blockIdx.x * blockDim.x + threadIdx.x;
    if (idx >= B_ * H_) return;
    int m = idx / H_, j = idx % H_;
    const float* g = gates + (size_t)m * G_;
    float ig = sigmoidf_(g[j]);
    float fg = sigmoidf_(g[H_ + j]);
    float gg = tanhf(g[2 * H_ + j]);
    float og = sigmoidf_(g[3 * H_ + j]);
    float c = fg * c_enc[idx] + ig * gg;
    float h = og * tanhf(c);
    c_enc[idx] = c;
    h_enc[idx] = h;
    att[(size_t)m * KENC + 2304 + j] = (bf16_t)h;
}
// variational head + rsample + KL; writes z into dec_in[:, 256:266]
__global__ void k_var_sample(const float* __restrict__ h_enc, const float* __restrict__ W_var,
                             const float* __restrict__ b_var, const float* __restrict__ eps_t,
                             bf16_t* __restrict__ dec_in, float* __restrict__ kl) {
    int idx = blockIdx.x * blockDim.x + threadIdx.x;
    if (idx >= B_ * Z_) return;
    int m = idx / Z_, j = idx % Z_;
    const float* h   = h_enc + (size_t)m * H_;
    const float* wmu = W_var + (size_t)j * H_;
    const float* wls = W_var + (size_t)(Z_ + j) * H_;
    float mu = b_var[j], ls = b_var[Z_ + j];
    for (int k = 0; k < H_; ++k) { float hv = h[k]; mu += hv * wmu[k]; ls += hv * wls[k]; }
    float sd = __expf(ls);
    float z  = mu + sd * eps_t[idx];
    kl[idx] += -ls + 0.5f * (sd * sd + mu * mu) - 0.5f;
    dec_in[(size_t)m * KDEC + H_ + j] = (bf16_t)z;
}
// decoder LSTM cell: updates c_dec; writes h_dec into dec_in[:,0:256] and att[:,2048:2304]
__global__ void k_dec_cell(const float* __restrict__ gates, float* __restrict__ c_dec,
                           bf16_t* __restrict__ dec_in, bf16_t* __restrict__ att) {
    int idx = blockIdx.x * blockDim.x + threadIdx.x;
    if (idx >= B_ * H_) return;
    int m = idx / H_, j = idx % H_;
    const float* g = gates + (size_t)m * G_;
    float ig = sigmoidf_(g[j]);
    float fg = sigmoidf_(g[H_ + j]);
    float gg = tanhf(g[2 * H_ + j]);
    float og = sigmoidf_(g[3 * H_ + j]);
    float c = fg * c_dec[idx] + ig * gg;
    float h = og * tanhf(c);
    c_dec[idx] = c;
    bf16_t hb = (bf16_t)h;
    dec_in[(size_t)m * KDEC + j] = hb;
    att[(size_t)m * KENC + 2048 + j] = hb;
}
__global__ void k_copy_f32(const float* __restrict__ src, float* __restrict__ dst, int n) {
    int i = blockIdx.x * blockDim.x + threadIdx.x;
    if (i < n) dst[i] = src[i];
}

// ---------------------------------------------------------------------------
extern "C" void kernel_launch(void* const* d_in, const int* in_sizes, int n_in,
                              void* d_out, int out_size, void* d_ws, size_t ws_size,
                              hipStream_t stream)
{
    const float* x        = (const float*)d_in[0];
    const float* eps      = (const float*)d_in[1];   // (T,B,Z)
    const float* W_ih_enc = (const float*)d_in[2];
    const float* b_ih_enc = (const float*)d_in[3];
    const float* W_hh_enc = (const float*)d_in[4];
    const float* b_hh_enc = (const float*)d_in[5];
    const float* W_ih_dec = (const float*)d_in[6];
    const float* b_ih_dec = (const float*)d_in[7];
    const float* W_hh_dec = (const float*)d_in[8];
    const float* b_hh_dec = (const float*)d_in[9];
    const float* W_var    = (const float*)d_in[10];
    const float* b_var    = (const float*)d_in[11];
    const float* W_write  = (const float*)d_in[12];
    const float* b_write  = (const float*)d_in[13];
    const float* W_obs    = (const float*)d_in[14];
    const float* b_obs    = (const float*)d_in[15];

    char* cur = (char*)d_ws;
    auto alloc = [&](size_t bytes) -> char* {
        char* p = cur;
        cur += (bytes + 255) & ~(size_t)255;
        return p;
    };

    bf16_t* wenc      = (bf16_t*)alloc((size_t)G_ * KENC * 2);
    bf16_t* wdec      = (bf16_t*)alloc((size_t)G_ * KDEC * 2);
    bf16_t* wwrite    = (bf16_t*)alloc((size_t)X_ * H_ * 2);
    bf16_t* wobs      = (bf16_t*)alloc((size_t)X_ * X_ * 2);
    float*  bias_enc  = (float*)alloc((size_t)G_ * 4);
    float*  bias_dec  = (float*)alloc((size_t)G_ * 4);
    bf16_t* att       = (bf16_t*)alloc((size_t)B_ * KENC * 2);
    bf16_t* dec_in    = (bf16_t*)alloc((size_t)B_ * KDEC * 2);
    float*  gates     = (float*)alloc((size_t)B_ * G_ * 4);
    float*  c_enc     = (float*)alloc((size_t)B_ * H_ * 4);
    float*  c_dec     = (float*)alloc((size_t)B_ * H_ * 4);
    float*  h_enc     = (float*)alloc((size_t)B_ * H_ * 4);
    float*  canvas    = (float*)alloc((size_t)B_ * X_ * 4);
    bf16_t* canvas_bf = (bf16_t*)alloc((size_t)B_ * X_ * 2);
    float*  kl        = (float*)alloc((size_t)B_ * Z_ * 4);

    float* x_mu_out = (float*)d_out;                 // B*X
    float* kl_out   = (float*)d_out + (size_t)B_ * X_;

    const int TPB = 256;
    auto nb = [](int n) { return (n + 255) / 256; };

    // --- one-time (per call) prep: bf16 weights, biases, state init -------
    k_build_wenc<<<nb(G_ * KENC), TPB, 0, stream>>>(W_ih_enc, W_hh_enc, wenc);
    k_build_wdec<<<nb(G_ * KDEC), TPB, 0, stream>>>(W_hh_dec, W_ih_dec, wdec);
    k_f32_to_bf16<<<nb(X_ * H_), TPB, 0, stream>>>(W_write, wwrite, X_ * H_);
    k_f32_to_bf16<<<nb(X_ * X_), TPB, 0, stream>>>(W_obs, wobs, X_ * X_);
    k_add_f32<<<nb(G_), TPB, 0, stream>>>(b_ih_enc, b_hh_enc, bias_enc, G_);
    k_add_f32<<<nb(G_), TPB, 0, stream>>>(b_ih_dec, b_hh_dec, bias_dec, G_);
    k_init_att<<<nb(B_ * KENC), TPB, 0, stream>>>(x, att);
    k_zero_bf16<<<nb(B_ * KDEC), TPB, 0, stream>>>(dec_in, B_ * KDEC);
    k_zero_f32<<<nb(B_ * H_), TPB, 0, stream>>>(c_enc, B_ * H_);
    k_zero_f32<<<nb(B_ * H_), TPB, 0, stream>>>(c_dec, B_ * H_);
    k_zero_f32<<<nb(B_ * X_), TPB, 0, stream>>>(canvas, B_ * X_);
    k_zero_f32<<<nb(B_ * Z_), TPB, 0, stream>>>(kl, B_ * Z_);

    dim3 gemm_grid(G_ / TN, B_ / TM);   // (16, 16): N=1024 cols, M=4096 rows

    // --- T sequential steps ------------------------------------------------
    for (int t = 0; t < T_; ++t) {
        const float* eps_t = eps + (size_t)t * B_ * Z_;

        // att[:,1024:2048] = x - sigmoid(canvas)
        k_att_xhat<<<nb(B_ * X_), TPB, 0, stream>>>(x, canvas, att);

        // encoder gates = att @ Wenc^T + (b_ih+b_hh)   (K=2560)
        k_gemm_bf16<KENC, false><<<gemm_grid, TPB, 0, stream>>>(
            att, KENC, wenc, KENC, gates, G_, bias_enc);
        k_enc_cell<<<nb(B_ * H_), TPB, 0, stream>>>(gates, c_enc, h_enc, att);

        // variational head + rsample + KL (writes z into dec_in)
        k_var_sample<<<nb(B_ * Z_), TPB, 0, stream>>>(h_enc, W_var, b_var, eps_t,
                                                      dec_in, kl);

        // decoder gates = [h_dec | z] @ Wdec^T + (b_ih+b_hh)   (K=288)
        k_gemm_bf16<KDEC, false><<<gemm_grid, TPB, 0, stream>>>(
            dec_in, KDEC, wdec, KDEC, gates, G_, bias_dec);
        k_dec_cell<<<nb(B_ * H_), TPB, 0, stream>>>(gates, c_dec, dec_in, att);

        // canvas += h_dec @ W_write^T + b_write   (K=256, uses dec_in[:,0:256])
        k_gemm_bf16<H_, true><<<gemm_grid, TPB, 0, stream>>>(
            dec_in, KDEC, wwrite, H_, canvas, X_, b_write);
    }

    // --- observation: x_mu = canvas @ W_obs^T + b_obs ----------------------
    k_f32_to_bf16<<<nb(B_ * X_), TPB, 0, stream>>>(canvas, canvas_bf, B_ * X_);
    k_gemm_bf16<X_, false><<<gemm_grid, TPB, 0, stream>>>(
        canvas_bf, X_, wobs, X_, x_mu_out, X_, b_obs);
    k_copy_f32<<<nb(B_ * Z_), TPB, 0, stream>>>(kl, kl_out, B_ * Z_);
}
